// MDTAttention_17008070492735
// MI455X (gfx1250) — matrix operand order repaired and checked by reading, not compiled, not run on hardware
//
#include <hip/hip_runtime.h>
#include <hip/hip_bf16.h>

// Problem constants
#define BDIM 8
#define CDIM 192
#define C3   576
#define NHEADS 4
#define HC   48
#define HDIM 128
#define WDIM 128
#define NTOT 16384            // H*W
#define KDIM 192              // inner dim of both 1x1-conv GEMMs
#define CHW  3145728          // C*H*W

typedef __attribute__((ext_vector_type(16))) __bf16 bf16x16;
typedef __attribute__((ext_vector_type(8)))  __bf16 bf16x8;
typedef __attribute__((ext_vector_type(8)))  float  f32x8;

#if __has_builtin(__builtin_amdgcn_global_load_async_to_lds_b128)
#define HAVE_ASYNC_LDS 1
typedef __attribute__((ext_vector_type(4))) int v4i_t;
typedef __attribute__((address_space(1))) v4i_t as1_v4i;   // global int4
typedef __attribute__((address_space(3))) v4i_t as3_v4i;   // LDS int4
#else
#define HAVE_ASYNC_LDS 0
#endif

union BF16Frag { bf16x16 v; bf16x8 h[2]; };

// ---- WMMA fragment loaders (CDNA5 ISA 7.12.2 layouts, wave32) ----
// A 16x32 bf16 from LDS [m][k]; row stride elements (stride*2 bytes mult of 16).
__device__ __forceinline__ bf16x16 load_frag_a(const __bf16* smem, int row0, int kk,
                                               int stride, int lane) {
  // lane<16: M=lane,    K = kk   + {0..7} then {16..23}
  // lane>=16: M=lane-16, K = kk+8 + {0..7} then {16..23}
  const __bf16* p = smem + (size_t)(row0 + (lane & 15)) * stride + kk + ((lane >> 4) << 3);
  BF16Frag f;
  f.h[0] = *(const bf16x8*)(p);
  f.h[1] = *(const bf16x8*)(p + 16);
  return f.v;
}
// B 32x16 bf16; LDS holds B transposed as [n][k].
__device__ __forceinline__ bf16x16 load_frag_b(const __bf16* smem, int col0, int kk,
                                               int stride, int lane) {
  // lane<16: N=lane, K = kk + {0..7} then {16..23} ; lane>=16: N=lane-16, K = kk+8 + {0..7} then {16..23}
  const __bf16* p = smem + (size_t)(col0 + (lane & 15)) * stride + kk + ((lane >> 4) << 3);
  BF16Frag f;
  f.h[0] = *(const bf16x8*)(p);
  f.h[1] = *(const bf16x8*)(p + 16);
  return f.v;
}
__device__ __forceinline__ f32x8 wmma_bf16(bf16x16 a, bf16x16 b, f32x8 c) {
  return __builtin_amdgcn_wmma_f32_16x16x32_bf16(false, a, false, b, (short)0, c, false, false);
}

// ---------------- Kernel 1: LN partial sums ----------------
__global__ __launch_bounds__(256) void k_ln_part(const float* __restrict__ x,
                                                 float* __restrict__ part) {
  const int chunk = blockIdx.x, b = blockIdx.y, tid = threadIdx.x;
  const size_t base = (size_t)b * CHW + (size_t)chunk * 16384;
  float s = 0.f, sq = 0.f;
  for (int i = tid; i < 16384; i += 256) {
    float v = x[base + i];
    s += v; sq += v * v;
  }
  __shared__ float r1[256], r2[256];
  r1[tid] = s; r2[tid] = sq; __syncthreads();
  for (int off = 128; off > 0; off >>= 1) {
    if (tid < off) { r1[tid] += r1[tid + off]; r2[tid] += r2[tid + off]; }
    __syncthreads();
  }
  if (tid == 0) {
    part[(b * 192 + chunk) * 2 + 0] = r1[0];
    part[(b * 192 + chunk) * 2 + 1] = r2[0];
  }
}

// ---------------- Kernel 2: LN finalize ----------------
__global__ __launch_bounds__(256) void k_ln_fin(const float* __restrict__ part,
                                                float* __restrict__ stats) {
  const int b = blockIdx.x, tid = threadIdx.x;
  float s = 0.f, sq = 0.f;
  for (int i = tid; i < 192; i += 256) {
    s  += part[(b * 192 + i) * 2 + 0];
    sq += part[(b * 192 + i) * 2 + 1];
  }
  __shared__ float r1[256], r2[256];
  r1[tid] = s; r2[tid] = sq; __syncthreads();
  for (int off = 128; off > 0; off >>= 1) {
    if (tid < off) { r1[tid] += r1[tid + off]; r2[tid] += r2[tid + off]; }
    __syncthreads();
  }
  if (tid == 0) {
    float mean = r1[0] / (float)CHW;
    float var  = r2[0] / (float)CHW - mean * mean;
    stats[b * 2 + 0] = mean;
    stats[b * 2 + 1] = rsqrtf(var + 1e-5f);
  }
}

// ---------------- Kernel 3: LN apply (float4) ----------------
__global__ __launch_bounds__(256) void k_ln_apply(const float* __restrict__ x,
                                                  const float* __restrict__ lw,
                                                  const float* __restrict__ lb,
                                                  const float* __restrict__ stats,
                                                  float* __restrict__ xn) {
  const size_t gid = (size_t)blockIdx.x * 256 + threadIdx.x;
  const size_t e = gid * 4;
  const int b = (int)(e / CHW);
  const size_t chw = e - (size_t)b * CHW;
  const float mean = stats[b * 2 + 0], rstd = stats[b * 2 + 1];
  float4 xv = ((const float4*)x)[gid];
  float4 wv = ((const float4*)lw)[chw >> 2];
  float4 bv = ((const float4*)lb)[chw >> 2];
  float4 o;
  o.x = (xv.x - mean) * rstd * wv.x + bv.x;
  o.y = (xv.y - mean) * rstd * wv.y + bv.y;
  o.z = (xv.z - mean) * rstd * wv.z + bv.z;
  o.w = (xv.w - mean) * rstd * wv.w + bv.w;
  ((float4*)xn)[gid] = o;
}

// ------- Kernel 4/9: WMMA GEMM  Out[b][m][n] = (resid) + bias[m] + sum_k A[m][k]*B[b][k][n]
// grid: (NTOT/128, M/64, B), block 256 (8 waves). K = 192. Block tile 64(M) x 128(N).
template <typename BT, typename OutT, bool HAS_RESID>
__global__ __launch_bounds__(256) void k_gemm(const float* __restrict__ A, size_t aBatchStride,
                                              const BT* __restrict__ Bsrc, size_t bBatchStride,
                                              const float* __restrict__ bias,
                                              const float* __restrict__ resid, size_t rBatchStride,
                                              OutT* __restrict__ Out, size_t oBatchStride) {
  const int b  = blockIdx.z;
  const int n0 = blockIdx.x * 128;
  const int m0 = blockIdx.y * 64;
  const float* Ag = A + (size_t)b * aBatchStride + (size_t)m0 * KDIM;
  const BT* Bg = Bsrc + (size_t)b * bBatchStride;
  __builtin_prefetch(Ag, 0, 1);  // global_prefetch_b8

  __shared__ __align__(16) __bf16 As[64][208];    // [m][k], 416B row stride
  __shared__ __align__(16) __bf16 Bs[128][208];   // [n][k] (B transposed)

  const int tid = threadIdx.x;
  for (int idx = tid; idx < 64 * KDIM; idx += 256) {
    int r = idx / KDIM, c = idx - r * KDIM;
    As[r][c] = (__bf16)Ag[(size_t)r * KDIM + c];
  }
  for (int idx = tid; idx < KDIM * 128; idx += 256) {
    int c = idx >> 7, j = idx & 127;
    Bs[j][c] = (__bf16)Bg[(size_t)c * NTOT + n0 + j];
  }
  __syncthreads();

  const int wave = tid >> 5, lane = tid & 31;
  const int tm  = (wave & 3) * 16;      // 4 waves along M
  const int tnb = (wave >> 2) * 64;     // 2 waves along N, 4 tiles each
  f32x8 acc[4] = {{}, {}, {}, {}};
  for (int kk = 0; kk < KDIM; kk += 32) {
    bf16x16 a = load_frag_a(&As[0][0], tm, kk, 208, lane);
#pragma unroll
    for (int t = 0; t < 4; ++t) {
      bf16x16 bb = load_frag_b(&Bs[0][0], tnb + t * 16, kk, 208, lane);
      acc[t] = wmma_bf16(a, bb, acc[t]);
    }
  }
  // Epilogue: D layout — VGPR r holds M = r + 8*(lane>=16), N = lane&15
  const int ln = lane & 15, lr = (lane >> 4) * 8;
  OutT* Og = Out + (size_t)b * oBatchStride;
  const float* Rg = HAS_RESID ? (resid + (size_t)b * rBatchStride) : nullptr;
#pragma unroll
  for (int t = 0; t < 4; ++t) {
#pragma unroll
    for (int r = 0; r < 8; ++r) {
      int mm = m0 + tm + lr + r;
      int nn = n0 + tnb + t * 16 + ln;
      float v = acc[t][r] + bias[mm];
      if (HAS_RESID) v += Rg[(size_t)mm * NTOT + nn];
      Og[(size_t)mm * NTOT + nn] = (OutT)v;
    }
  }
}

// ------- Kernel 5: grouped 3x3 conv (groups=C, 3in/3out) + q/k sumsq partials
__global__ __launch_bounds__(256) void k_dw(const __bf16* __restrict__ qkv,
                                            const float* __restrict__ dww,
                                            const float* __restrict__ dwb,
                                            __bf16* __restrict__ dwout,
                                            float* __restrict__ npart) {
  const int band = blockIdx.x, g = blockIdx.y, b = blockIdx.z, tid = threadIdx.x;
  __shared__ float tile[3][18][132];   // rows band*16-1..+16, cols -1..128
  __shared__ float wlds[81];
  __shared__ float red[256];

  if (tid < 81) wlds[tid] = dww[(size_t)g * 81 + tid];
  const size_t inbase = ((size_t)b * C3 + g * 3) * NTOT;
  for (int idx = tid; idx < 3 * 18 * 130; idx += 256) {
    int ci = idx / (18 * 130);
    int rem = idx - ci * (18 * 130);
    int y = rem / 130, xx = rem - y * 130;
    int gy = band * 16 + y - 1, gx = xx - 1;
    float v = 0.f;
    if (gy >= 0 && gy < HDIM && gx >= 0 && gx < WDIM)
      v = (float)qkv[inbase + (size_t)ci * NTOT + gy * WDIM + gx];
    tile[ci][y][xx] = v;
  }
  __syncthreads();

  float sq0 = 0.f, sq1 = 0.f, sq2 = 0.f;
  const float bia0 = dwb[g * 3 + 0], bia1 = dwb[g * 3 + 1], bia2 = dwb[g * 3 + 2];
  const size_t outbase = ((size_t)b * C3 + g * 3) * NTOT;
  for (int p = tid; p < 16 * WDIM; p += 256) {
    int y = p >> 7, xx = p & 127;
    float o0 = bia0, o1 = bia1, o2 = bia2;
    for (int ci = 0; ci < 3; ++ci)
      for (int ky = 0; ky < 3; ++ky)
        for (int kx = 0; kx < 3; ++kx) {
          float v = tile[ci][y + ky][xx + kx];
          int wi = ci * 9 + ky * 3 + kx;
          o0 += wlds[wi] * v;
          o1 += wlds[27 + wi] * v;
          o2 += wlds[54 + wi] * v;
        }
    size_t pix = (size_t)(band * 16 + y) * WDIM + xx;
    dwout[outbase + 0 * NTOT + pix] = (__bf16)o0;
    dwout[outbase + 1 * NTOT + pix] = (__bf16)o1;
    dwout[outbase + 2 * NTOT + pix] = (__bf16)o2;
    sq0 += o0 * o0; sq1 += o1 * o1; sq2 += o2 * o2;
  }
  if (g < 128) {  // only q (g<64) and k (64<=g<128) need norms
    float sqs[3] = {sq0, sq1, sq2};
    for (int j = 0; j < 3; ++j) {
      __syncthreads();
      red[tid] = sqs[j];
      __syncthreads();
      for (int off = 128; off > 0; off >>= 1) {
        if (tid < off) red[tid] += red[tid + off];
        __syncthreads();
      }
      if (tid == 0)
        npart[((size_t)b * 384 + g * 3 + j) * 8 + band] = red[0];
    }
  }
}

// ---------------- Kernel 6: finalize 1/max(||.||, eps) ----------------
__global__ __launch_bounds__(256) void k_norm(const float* __restrict__ npart,
                                              float* __restrict__ rnorm) {
  int gid = blockIdx.x * 256 + threadIdx.x;
  if (gid < BDIM * 384) {
    float s = 0.f;
    for (int i = 0; i < 8; ++i) s += npart[(size_t)gid * 8 + i];
    float nrm = sqrtf(s);
    rnorm[gid] = 1.f / fmaxf(nrm, 1e-12f);
  }
}

// ---- async (or fallback) staging of one 48x128 bf16 chunk into LDS ----
__device__ __forceinline__ void stage_chunk(const __bf16* qg, const __bf16* kg, int kc,
                                            __bf16 (*qbuf)[136], __bf16 (*kbuf)[136], int tid) {
  for (int idx = tid; idx < 48 * 16; idx += 256) {
    int ch = idx >> 4, seg = idx & 15;
    const __bf16* gq = qg + (size_t)ch * NTOT + kc + seg * 8;
    const __bf16* gk = kg + (size_t)ch * NTOT + kc + seg * 8;
#if HAVE_ASYNC_LDS
    __builtin_amdgcn_global_load_async_to_lds_b128(
        (as1_v4i*)(__bf16*)gq, (as3_v4i*)&qbuf[ch][seg * 8], 0, 0);
    __builtin_amdgcn_global_load_async_to_lds_b128(
        (as1_v4i*)(__bf16*)gk, (as3_v4i*)&kbuf[ch][seg * 8], 0, 0);
#else
    *(bf16x8*)&qbuf[ch][seg * 8] = *(const bf16x8*)gq;
    *(bf16x8*)&kbuf[ch][seg * 8] = *(const bf16x8*)gk;
#endif
  }
}
__device__ __forceinline__ void wait_async_lds() {
#if HAVE_ASYNC_LDS
#if __has_builtin(__builtin_amdgcn_s_wait_asynccnt)
  __builtin_amdgcn_s_wait_asynccnt(0);
#else
  asm volatile("s_wait_asynccnt 0x0" ::: "memory");
#endif
#endif
}

// ------- Kernel 7: attention G = (q/|q|)·(k/|k|)^T / scale, softmax rows -------
// grid (NH, B), block 256 (8 waves). 48x48 output (3x3 tiles of 16x16), K=16384 in
// 128-chunks, async double-buffered. Static accumulator mapping: acc0 = tile 'wave',
// acc1 = tile 8 (all waves compute it; only wave 0 writes it) -> no indirect VGPR moves.
__global__ __launch_bounds__(256) void k_attn(const __bf16* __restrict__ dwout,
                                              const float* __restrict__ rnorm,
                                              const float* __restrict__ scalep,
                                              float* __restrict__ Smat) {
  const int nh = blockIdx.x, b = blockIdx.y, tid = threadIdx.x;
  const int wave = tid >> 5, lane = tid & 31;
  const float inv_scale = 1.0f / scalep[0];
  const __bf16* qg = dwout + ((size_t)b * C3 + nh * HC) * NTOT;
  const __bf16* kg = dwout + ((size_t)b * C3 + CDIM + nh * HC) * NTOT;

  __shared__ __align__(16) __bf16 qs[2][48][136];   // double buffered
  __shared__ __align__(16) __bf16 ks[2][48][136];
  __shared__ float Gsh[48][49];

  const int mi0 = (wave / 3) * 16, nj0 = (wave % 3) * 16;  // tile 'wave'
  f32x8 acc0 = {};    // tile 'wave'
  f32x8 acc1 = {};    // tile 8 at (32,32), redundantly computed by all waves
  const int NCH = NTOT / 128;

  stage_chunk(qg, kg, 0, qs[0], ks[0], tid);
  wait_async_lds();
  __syncthreads();

  for (int i = 0; i < NCH; ++i) {
    const int cur = i & 1;
    if (i + 1 < NCH)
      stage_chunk(qg, kg, (i + 1) * 128, qs[cur ^ 1], ks[cur ^ 1], tid);
#pragma unroll
    for (int kk = 0; kk < 128; kk += 32) {
      bf16x16 a0 = load_frag_a(&qs[cur][0][0], mi0, kk, 136, lane);
      bf16x16 b0 = load_frag_b(&ks[cur][0][0], nj0, kk, 136, lane);
      acc0 = wmma_bf16(a0, b0, acc0);
      bf16x16 a1 = load_frag_a(&qs[cur][0][0], 32, kk, 136, lane);
      bf16x16 b1 = load_frag_b(&ks[cur][0][0], 32, kk, 136, lane);
      acc1 = wmma_bf16(a1, b1, acc1);
    }
    wait_async_lds();
    __syncthreads();
  }

  // scale by row/col rnorm and 1/scale, stash to LDS
  {
    int jj = nj0 + (lane & 15);
    float rk = rnorm[b * 384 + CDIM + nh * HC + jj];
#pragma unroll
    for (int r = 0; r < 8; ++r) {
      int ii = mi0 + ((lane >> 4) * 8) + r;
      float rq = rnorm[b * 384 + nh * HC + ii];
      Gsh[ii][jj] = acc0[r] * rq * rk * inv_scale;
    }
  }
  if (wave == 0) {   // tile 8 at (32,32)
    int jj = 32 + (lane & 15);
    float rk = rnorm[b * 384 + CDIM + nh * HC + jj];
#pragma unroll
    for (int r = 0; r < 8; ++r) {
      int ii = 32 + ((lane >> 4) * 8) + r;
      float rq = rnorm[b * 384 + nh * HC + ii];
      Gsh[ii][jj] = acc1[r] * rq * rk * inv_scale;
    }
  }
  __syncthreads();
  // softmax over columns j within each row i (two-pass)
  if (tid < 48) {
    float mx = -3.0e38f;
    for (int j = 0; j < 48; ++j) mx = fmaxf(mx, Gsh[tid][j]);
    float s = 0.f;
    for (int j = 0; j < 48; ++j) s += __expf(Gsh[tid][j] - mx);
    float inv = 1.f / s;
    float* srow = Smat + (((size_t)(b * NHEADS + nh) * HC + tid) * HC);
    for (int j = 0; j < 48; ++j) srow[j] = __expf(Gsh[tid][j] - mx) * inv;
  }
}

// ------- Kernel 8: M_b[o][head*48+i] = sum_j Wout[o][head*48+j] * S[b,head,i,j]
__global__ __launch_bounds__(256) void k_buildM(const float* __restrict__ Wout,
                                                const float* __restrict__ Smat,
                                                float* __restrict__ Mmat) {
  int gid = blockIdx.x * 256 + threadIdx.x;
  if (gid < BDIM * CDIM * CDIM) {
    int b = gid / (CDIM * CDIM);
    int rem = gid - b * (CDIM * CDIM);
    int o = rem / CDIM;
    int k2 = rem - o * CDIM;
    int head = k2 / HC, i = k2 - head * HC;
    const float* wrow = Wout + (size_t)o * CDIM + head * HC;
    const float* srow = Smat + (((size_t)(b * NHEADS + head) * HC + i) * HC);
    float acc = 0.f;
    for (int j = 0; j < HC; ++j) acc += wrow[j] * srow[j];
    Mmat[gid] = acc;
  }
}

extern "C" void kernel_launch(void* const* d_in, const int* in_sizes, int n_in,
                              void* d_out, int out_size, void* d_ws, size_t ws_size,
                              hipStream_t stream) {
  const float* x     = (const float*)d_in[0];
  const float* ln_w  = (const float*)d_in[1];
  const float* ln_b  = (const float*)d_in[2];
  const float* qkv_w = (const float*)d_in[3];
  const float* qkv_b = (const float*)d_in[4];
  const float* dw_w  = (const float*)d_in[5];
  const float* dw_b  = (const float*)d_in[6];
  const float* scale = (const float*)d_in[7];
  const float* out_w = (const float*)d_in[8];
  const float* out_b = (const float*)d_in[9];
  float* out = (float*)d_out;

  // Workspace carve-up
  char* wsb = (char*)d_ws;
  float*  part   = (float*)wsb;  wsb += (size_t)3072 * 4;
  float*  stats  = (float*)wsb;  wsb += (size_t)16 * 4;
  float*  xn     = (float*)wsb;  wsb += (size_t)25165824 * 4;   // B*C*N f32
  __bf16* qkvb   = (__bf16*)wsb; wsb += (size_t)75497472 * 2;   // B*3C*N bf16
  __bf16* dwoutb = (__bf16*)wsb; wsb += (size_t)75497472 * 2;   // B*3C*N bf16
  float*  npart  = (float*)wsb;  wsb += (size_t)24576 * 4;
  float*  rnorm  = (float*)wsb;  wsb += (size_t)3072 * 4;
  float*  Smat   = (float*)wsb;  wsb += (size_t)73728 * 4;
  float*  Mmat   = (float*)wsb;

  // 1-3: LayerNorm
  k_ln_part<<<dim3(192, BDIM), 256, 0, stream>>>(x, part);
  k_ln_fin<<<BDIM, 256, 0, stream>>>(part, stats);
  k_ln_apply<<<(BDIM * CHW) / 1024, 256, 0, stream>>>(x, ln_w, ln_b, stats, xn);

  // 4: qkv = W_qkv @ xn + b  (WMMA GEMM, M=576, bf16 out)
  k_gemm<float, __bf16, false><<<dim3(NTOT / 128, C3 / 64, BDIM), 256, 0, stream>>>(
      qkv_w, (size_t)0, xn, (size_t)CDIM * NTOT, qkv_b,
      nullptr, (size_t)0, qkvb, (size_t)C3 * NTOT);

  // 5: grouped 3x3 conv + q/k sumsq partials (bf16 in/out, f32 accum)
  k_dw<<<dim3(8, CDIM, BDIM), 256, 0, stream>>>(qkvb, dw_w, dw_b, dwoutb, npart);

  // 6: reciprocal norms
  k_norm<<<(BDIM * 384 + 255) / 256, 256, 0, stream>>>(npart, rnorm);

  // 7: 48x48 attention + softmax (WMMA over K=16384, async double-buffered LDS)
  k_attn<<<dim3(NHEADS, BDIM), 256, 0, stream>>>(dwoutb, rnorm, scale, Smat);

  // 8: fused projection matrix M_b = Wout · blockdiag(S^T)
  k_buildM<<<(BDIM * CDIM * CDIM + 255) / 256, 256, 0, stream>>>(out_w, Smat, Mmat);

  // 9: out = xn + M_b @ v + b_out  (WMMA GEMM, M=192, residual, bf16 B)
  k_gemm<__bf16, float, true><<<dim3(NTOT / 128, CDIM / 64, BDIM), 256, 0, stream>>>(
      Mmat, (size_t)(CDIM * CDIM), dwoutb + (size_t)384 * NTOT, (size_t)C3 * NTOT, out_b,
      xn, (size_t)CDIM * NTOT, out, (size_t)CDIM * NTOT);
}
